// FunnelEncoderLayer_67894843015325
// MI455X (gfx1250) — compile-verified, matrix operand order checked by reading, not run on hardware
//
#include <hip/hip_runtime.h>
#include <math.h>

// ---------------- problem constants ----------------
namespace {
constexpr int BB = 32, SS = 512, DMODEL = 512, NHEAD = 8, DKV = 64, DINNER = 2048;
constexpr int MROWS = BB * SS;                       // 16384 token rows
constexpr long ENC_ELEMS = (long)BB * SS * DMODEL;   // 8388608 floats (enc_out)
}

typedef __attribute__((ext_vector_type(16))) _Float16 v16h;
typedef __attribute__((ext_vector_type(8)))  _Float16 v8h;
typedef __attribute__((ext_vector_type(8)))  float    v8f;

// ---- WMMA fragment loaders (CDNA5 16-bit A/B layout) ----
// lane<16  : K = k0+0..7  and k0+16..23  of its row/column
// lane>=16 : K = k0+8..15 and k0+24..31
__device__ __forceinline__ v16h frag_f16(const _Float16* rowk0, int lhi) {
  const _Float16* p = rowk0 + (lhi ? 8 : 0);
  v8h a = *(const v8h*)p;
  v8h b = *(const v8h*)(p + 16);
  v16h r;
#pragma unroll
  for (int i = 0; i < 8; ++i) { r[i] = a[i]; r[i + 8] = b[i]; }
  return r;
}
// raw fp32 fragment (kept in fp32 through the pipeline; converted at WMMA issue)
struct rawA { v8f lo, hi; };
__device__ __forceinline__ rawA raw_f32(const float* rowk0, int lhi) {
  const float* q = rowk0 + (lhi ? 8 : 0);
  rawA r;
  r.lo = *(const v8f*)q;
  r.hi = *(const v8f*)(q + 16);
  return r;
}
__device__ __forceinline__ v16h cvtA(rawA r) {
  v16h o;
#pragma unroll
  for (int i = 0; i < 8; ++i) { o[i] = (_Float16)r.lo[i]; o[i + 8] = (_Float16)r.hi[i]; }
  return o;
}
#define WMMA(c, a, b) \
  c = __builtin_amdgcn_wmma_f32_16x16x32_f16(false, a, false, b, (short)0, c, false, false)

// C/D layout: VGPR r, lanes 0-15 -> (M=r, N=lane); lanes 16-31 -> (M=r+8, N=lane-16)

// ---- software-pipelined 16x64 strip GEMM (double-buffered fragments) ----
template <int KTOT>
__device__ __forceinline__ void mm_strip(const _Float16* __restrict__ arow,
                                         const _Float16* __restrict__ b0,
                                         const _Float16* __restrict__ b1,
                                         const _Float16* __restrict__ b2,
                                         const _Float16* __restrict__ b3, int lhi,
                                         v8f& c0, v8f& c1, v8f& c2, v8f& c3) {
  v16h a  = frag_f16(arow, lhi);
  v16h f0 = frag_f16(b0, lhi), f1 = frag_f16(b1, lhi);
  v16h f2 = frag_f16(b2, lhi), f3 = frag_f16(b3, lhi);
#pragma unroll
  for (int k0 = 32; k0 < KTOT; k0 += 32) {
    v16h an = frag_f16(arow + k0, lhi);            // issue next-iter loads first
    v16h g0 = frag_f16(b0 + k0, lhi);
    v16h g1 = frag_f16(b1 + k0, lhi);
    v16h g2 = frag_f16(b2 + k0, lhi);
    v16h g3 = frag_f16(b3 + k0, lhi);
    WMMA(c0, a, f0);                               // compute on prev-iter data
    WMMA(c1, a, f1);
    WMMA(c2, a, f2);
    WMMA(c3, a, f3);
    a = an; f0 = g0; f1 = g1; f2 = g2; f3 = g3;
  }
  WMMA(c0, a, f0);
  WMMA(c1, a, f1);
  WMMA(c2, a, f2);
  WMMA(c3, a, f3);
}

// ---------------- weight packing (one-time per launch) ----------------
// wt[(h*64+kk)*512 + d] = w_qs[h][d][kk]   (per-lane contiguous B frags)
__global__ void k_pack_qkv(const float* __restrict__ w, _Float16* __restrict__ wt) {
  int i = blockIdx.x * 256 + threadIdx.x;
  if (i >= DMODEL * DMODEL) return;
  int n = i / DMODEL, d = i % DMODEL;
  int h = n >> 6, kk = n & 63;
  wt[i] = (_Float16)w[(h * DMODEL + d) * DKV + kk];
}
// generic transpose: w is [K][N] row-major, wt[n*K + d] = w[d*N + n]
__global__ void k_pack_T(const float* __restrict__ w, _Float16* __restrict__ wt, int N, int K) {
  int i = blockIdx.x * 256 + threadIdx.x;
  if (i >= N * K) return;
  int n = i / K, d = i % K;
  wt[i] = (_Float16)w[(long)d * N + n];
}

// ---------------- skipper: stable compaction ----------------
__global__ void k_compact(const int* __restrict__ keep, int* __restrict__ srcidx,
                          int* __restrict__ nkeep) {
  int b = blockIdx.x, s = threadIdx.x;
  __shared__ int scan[SS];
  int f = keep[b * SS + s] > 0 ? 1 : 0;
  scan[s] = f;
  __syncthreads();
  for (int o = 1; o < SS; o <<= 1) {
    int v = (s >= o) ? scan[s - o] : 0;
    __syncthreads();
    scan[s] += v;
    __syncthreads();
  }
  int incl = scan[s];
  if (f) srcidx[b * SS + (incl - f)] = s;
  if (s == SS - 1) nkeep[b] = incl;
}

// gather kept rows to front, zero-pad; produce fp32 master + f16 shadow
__global__ void k_gather(const float* __restrict__ x, const int* __restrict__ srcidx,
                         const int* __restrict__ nkeep, float* __restrict__ xc,
                         _Float16* __restrict__ xch) {
  int b = blockIdx.x >> 9, i = blockIdx.x & 511;
  int nk = nkeep[b];
  const float* src = (i < nk) ? x + ((long)b * SS + srcidx[b * SS + i]) * DMODEL : nullptr;
  long ro = ((long)b * SS + i) * DMODEL;
  for (int d = threadIdx.x; d < DMODEL; d += 128) {
    float v = src ? src[d] : 0.0f;
    xc[ro + d] = v;
    xch[ro + d] = (_Float16)v;
  }
}

// ================= blocked WMMA GEMMs: one wave owns a 16x64 strip =================

// Q/K projection: [16384,512] x [512,512] -> f16 [row][col]
__global__ void k_gemm_qk(const _Float16* __restrict__ A, const _Float16* __restrict__ Bt,
                          _Float16* __restrict__ out) {
  int wave = (blockIdx.x * blockDim.x + threadIdx.x) >> 5;
  int lane = threadIdx.x & 31, l15 = lane & 15, lhi = lane >> 4;
  int n0 = (wave & 7) * 64;          // 8 strips of 64 cols
  int m0 = (wave >> 3) * 16;         // 1024 row tiles
  const _Float16* arow = A + (long)(m0 + l15) * DMODEL;
  const _Float16* b0 = Bt + (long)(n0 + 0  + l15) * DMODEL;
  const _Float16* b1 = Bt + (long)(n0 + 16 + l15) * DMODEL;
  const _Float16* b2 = Bt + (long)(n0 + 32 + l15) * DMODEL;
  const _Float16* b3 = Bt + (long)(n0 + 48 + l15) * DMODEL;
  v8f c0 = {}, c1 = {}, c2 = {}, c3 = {};
  mm_strip<DMODEL>(arow, b0, b1, b2, b3, lhi, c0, c1, c2, c3);
#pragma unroll
  for (int r = 0; r < 8; ++r) {
    long row = m0 + r + lhi * 8;
    out[row * DMODEL + n0 + 0  + l15] = (_Float16)c0[r];
    out[row * DMODEL + n0 + 16 + l15] = (_Float16)c1[r];
    out[row * DMODEL + n0 + 32 + l15] = (_Float16)c2[r];
    out[row * DMODEL + n0 + 48 + l15] = (_Float16)c3[r];
  }
}

// V projection with transposed store: vT[((b*8+h)*64+kk)*512 + t]
__global__ void k_gemm_v(const _Float16* __restrict__ A, const _Float16* __restrict__ Bt,
                         _Float16* __restrict__ vT) {
  int wave = (blockIdx.x * blockDim.x + threadIdx.x) >> 5;
  int lane = threadIdx.x & 31, l15 = lane & 15, lhi = lane >> 4;
  int n0 = (wave & 7) * 64;
  int m0 = (wave >> 3) * 16;
  const _Float16* arow = A + (long)(m0 + l15) * DMODEL;
  const _Float16* b0 = Bt + (long)(n0 + 0  + l15) * DMODEL;
  const _Float16* b1 = Bt + (long)(n0 + 16 + l15) * DMODEL;
  const _Float16* b2 = Bt + (long)(n0 + 32 + l15) * DMODEL;
  const _Float16* b3 = Bt + (long)(n0 + 48 + l15) * DMODEL;
  v8f c0 = {}, c1 = {}, c2 = {}, c3 = {};
  mm_strip<DMODEL>(arow, b0, b1, b2, b3, lhi, c0, c1, c2, c3);
#pragma unroll
  for (int r = 0; r < 8; ++r) {
    int row = m0 + r + lhi * 8;
    int b_ = row >> 9, t = row & 511;
    int h = n0 >> 6;                   // strip is 64-aligned -> single head
    long basev = ((long)(b_ * NHEAD + h) * DKV) * SS + t;
    vT[basev + (long)(((n0 + 0  + l15) & 63)) * SS] = (_Float16)c0[r];
    vT[basev + (long)(((n0 + 16 + l15) & 63)) * SS] = (_Float16)c1[r];
    vT[basev + (long)(((n0 + 32 + l15) & 63)) * SS] = (_Float16)c2[r];
    vT[basev + (long)(((n0 + 48 + l15) & 63)) * SS] = (_Float16)c3[r];
  }
}

// scores = Q Kt / sqrt(D_MODEL) -> attns[h,b,s,t]; one wave: 16 s-rows x 64 t-cols
__global__ void k_scores(const _Float16* __restrict__ q, const _Float16* __restrict__ k,
                         float* __restrict__ attns) {
  int wave = (blockIdx.x * blockDim.x + threadIdx.x) >> 5;
  int lane = threadIdx.x & 31, l15 = lane & 15, lhi = lane >> 4;
  int t0 = (wave & 7) * 64; wave >>= 3;     // 8 t-strips
  int s0 = (wave & 31) * 16; wave >>= 5;    // 32 s-tiles
  int h = wave & 7, b = wave >> 3;
  const _Float16* arow = q + ((long)(b * SS + s0 + l15) * DMODEL + h * DKV);
  const _Float16* b0 = k + ((long)(b * SS + t0 + 0  + l15) * DMODEL + h * DKV);
  const _Float16* b1 = k + ((long)(b * SS + t0 + 16 + l15) * DMODEL + h * DKV);
  const _Float16* b2 = k + ((long)(b * SS + t0 + 32 + l15) * DMODEL + h * DKV);
  const _Float16* b3 = k + ((long)(b * SS + t0 + 48 + l15) * DMODEL + h * DKV);
  v8f c0 = {}, c1 = {}, c2 = {}, c3 = {};
  mm_strip<DKV>(arow, b0, b1, b2, b3, lhi, c0, c1, c2, c3);
  const float scale = 0.04419417382415922f;  // 1/sqrt(512)
#pragma unroll
  for (int r = 0; r < 8; ++r) {
    int s = s0 + r + lhi * 8;
    float* p = attns + ((long)(h * BB + b) * SS + s) * SS;
    p[t0 + 0  + l15] = c0[r] * scale;
    p[t0 + 16 + l15] = c1[r] * scale;
    p[t0 + 32 + l15] = c2[r] * scale;
    p[t0 + 48 + l15] = c3[r] * scale;
  }
}

// ---------------- in-place masked softmax over last dim ----------------
__global__ void k_softmax(float* __restrict__ attns, const int* __restrict__ nkeep) {
  long row = blockIdx.x;                 // (h*32 + b)*512 + s
  int b = (int)((row >> 9) & 31);
  float* p = attns + row * SS;
  int nk = nkeep[b];
  int t = threadIdx.x;
  __shared__ float red[SS];
  float v = (t < nk) ? p[t] : -INFINITY;
  red[t] = v; __syncthreads();
  for (int o = 256; o > 0; o >>= 1) { if (t < o) red[t] = fmaxf(red[t], red[t + o]); __syncthreads(); }
  float m = red[0]; __syncthreads();
  float e = (t < nk) ? __expf(v - m) : 0.0f;
  red[t] = e; __syncthreads();
  for (int o = 256; o > 0; o >>= 1) { if (t < o) red[t] += red[t + o]; __syncthreads(); }
  p[t] = e / red[0];
}

// heads = attn @ V; one wave: 16 s-rows x all 64 head dims.
// A (fp32 probs) stays raw through the double buffer; converted at WMMA issue.
__global__ void k_attnv(const float* __restrict__ attns, const _Float16* __restrict__ vT,
                        _Float16* __restrict__ heads) {
  int wave = (blockIdx.x * blockDim.x + threadIdx.x) >> 5;
  int lane = threadIdx.x & 31, l15 = lane & 15, lhi = lane >> 4;
  int s0 = (wave & 31) * 16; wave >>= 5;
  int h = wave & 7, b = wave >> 3;
  const float*    arow = attns + ((long)(h * BB + b) * SS + s0 + l15) * SS;
  const _Float16* b0 = vT + ((long)(b * NHEAD + h) * DKV + 0  + l15) * SS;
  const _Float16* b1 = vT + ((long)(b * NHEAD + h) * DKV + 16 + l15) * SS;
  const _Float16* b2 = vT + ((long)(b * NHEAD + h) * DKV + 32 + l15) * SS;
  const _Float16* b3 = vT + ((long)(b * NHEAD + h) * DKV + 48 + l15) * SS;
  v8f c0 = {}, c1 = {}, c2 = {}, c3 = {};
  rawA ar = raw_f32(arow, lhi);
  v16h f0 = frag_f16(b0, lhi), f1 = frag_f16(b1, lhi);
  v16h f2 = frag_f16(b2, lhi), f3 = frag_f16(b3, lhi);
#pragma unroll
  for (int k0 = 32; k0 < SS; k0 += 32) {
    rawA an = raw_f32(arow + k0, lhi);
    v16h g0 = frag_f16(b0 + k0, lhi);
    v16h g1 = frag_f16(b1 + k0, lhi);
    v16h g2 = frag_f16(b2 + k0, lhi);
    v16h g3 = frag_f16(b3 + k0, lhi);
    __builtin_prefetch(arow + k0 + 32, 0, 1);     // global_prefetch_b8: HBM stream ahead
    v16h ah = cvtA(ar);
    WMMA(c0, ah, f0);
    WMMA(c1, ah, f1);
    WMMA(c2, ah, f2);
    WMMA(c3, ah, f3);
    ar = an; f0 = g0; f1 = g1; f2 = g2; f3 = g3;
  }
  v16h ah = cvtA(ar);
  WMMA(c0, ah, f0);
  WMMA(c1, ah, f1);
  WMMA(c2, ah, f2);
  WMMA(c3, ah, f3);
#pragma unroll
  for (int r = 0; r < 8; ++r) {
    int s = s0 + r + lhi * 8;
    _Float16* p = heads + ((long)(b * SS + s)) * DMODEL + h * DKV;
    p[0  + l15] = (_Float16)c0[r];
    p[16 + l15] = (_Float16)c1[r];
    p[32 + l15] = (_Float16)c2[r];
    p[48 + l15] = (_Float16)c3[r];
  }
}

// proj GEMM + bias + residual -> preLN fp32
__global__ void k_proj(const _Float16* __restrict__ heads, const _Float16* __restrict__ Bt,
                       const float* __restrict__ bias, const float* __restrict__ resid,
                       float* __restrict__ preLN) {
  int wave = (blockIdx.x * blockDim.x + threadIdx.x) >> 5;
  int lane = threadIdx.x & 31, l15 = lane & 15, lhi = lane >> 4;
  int n0 = (wave & 7) * 64;
  int m0 = (wave >> 3) * 16;
  const _Float16* arow = heads + (long)(m0 + l15) * DMODEL;
  const _Float16* b0 = Bt + (long)(n0 + 0  + l15) * DMODEL;
  const _Float16* b1 = Bt + (long)(n0 + 16 + l15) * DMODEL;
  const _Float16* b2 = Bt + (long)(n0 + 32 + l15) * DMODEL;
  const _Float16* b3 = Bt + (long)(n0 + 48 + l15) * DMODEL;
  v8f c0 = {}, c1 = {}, c2 = {}, c3 = {};
  mm_strip<DMODEL>(arow, b0, b1, b2, b3, lhi, c0, c1, c2, c3);
#pragma unroll
  for (int r = 0; r < 8; ++r) {
    long row = m0 + r + lhi * 8;
    long i0 = row * DMODEL + n0 + l15;
    preLN[i0 + 0 ] = c0[r] + bias[n0 + 0  + l15] + resid[i0 + 0 ];
    preLN[i0 + 16] = c1[r] + bias[n0 + 16 + l15] + resid[i0 + 16];
    preLN[i0 + 32] = c2[r] + bias[n0 + 32 + l15] + resid[i0 + 32];
    preLN[i0 + 48] = c3[r] + bias[n0 + 48 + l15] + resid[i0 + 48];
  }
}

// LayerNorm (unbiased var, eps added to std); optional f16 shadow output
__global__ void k_layernorm(const float* __restrict__ in, const float* __restrict__ g,
                            const float* __restrict__ bta, float* __restrict__ out,
                            _Float16* __restrict__ out16) {
  int row = blockIdx.x, t = threadIdx.x;
  __shared__ float red[DMODEL];
  float x = in[(long)row * DMODEL + t];
  red[t] = x; __syncthreads();
  for (int o = 256; o > 0; o >>= 1) { if (t < o) red[t] += red[t + o]; __syncthreads(); }
  float mu = red[0] * (1.0f / DMODEL); __syncthreads();
  float d = x - mu;
  red[t] = d * d; __syncthreads();
  for (int o = 256; o > 0; o >>= 1) { if (t < o) red[t] += red[t + o]; __syncthreads(); }
  float var = red[0] * (1.0f / (DMODEL - 1));
  float y = d / (sqrtf(var) + 1e-3f) * g[t] + bta[t];
  out[(long)row * DMODEL + t] = y;
  if (out16) out16[(long)row * DMODEL + t] = (_Float16)y;
}

// FFN1: relu(x@W1 + b1) -> f16   [16384,512]x[512,2048]
__global__ void k_ffn1(const _Float16* __restrict__ A, const _Float16* __restrict__ Bt,
                       const float* __restrict__ bias, _Float16* __restrict__ out) {
  int wave = (blockIdx.x * blockDim.x + threadIdx.x) >> 5;
  int lane = threadIdx.x & 31, l15 = lane & 15, lhi = lane >> 4;
  int n0 = (wave & 31) * 64;          // 32 strips of 64 over DINNER
  int m0 = (wave >> 5) * 16;
  const _Float16* arow = A + (long)(m0 + l15) * DMODEL;
  const _Float16* b0 = Bt + (long)(n0 + 0  + l15) * DMODEL;
  const _Float16* b1 = Bt + (long)(n0 + 16 + l15) * DMODEL;
  const _Float16* b2 = Bt + (long)(n0 + 32 + l15) * DMODEL;
  const _Float16* b3 = Bt + (long)(n0 + 48 + l15) * DMODEL;
  v8f c0 = {}, c1 = {}, c2 = {}, c3 = {};
  mm_strip<DMODEL>(arow, b0, b1, b2, b3, lhi, c0, c1, c2, c3);
#pragma unroll
  for (int r = 0; r < 8; ++r) {
    long row = m0 + r + lhi * 8;
    _Float16* p = out + row * DINNER + n0 + l15;
    p[0 ] = (_Float16)fmaxf(c0[r] + bias[n0 + 0  + l15], 0.0f);
    p[16] = (_Float16)fmaxf(c1[r] + bias[n0 + 16 + l15], 0.0f);
    p[32] = (_Float16)fmaxf(c2[r] + bias[n0 + 32 + l15], 0.0f);
    p[48] = (_Float16)fmaxf(c3[r] + bias[n0 + 48 + l15], 0.0f);
  }
}

// FFN2: h@W2 + b2 + resid -> preLN fp32   [16384,2048]x[2048,512]
__global__ void k_ffn2(const _Float16* __restrict__ A, const _Float16* __restrict__ Bt,
                       const float* __restrict__ bias, const float* __restrict__ resid,
                       float* __restrict__ preLN) {
  int wave = (blockIdx.x * blockDim.x + threadIdx.x) >> 5;
  int lane = threadIdx.x & 31, l15 = lane & 15, lhi = lane >> 4;
  int n0 = (wave & 7) * 64;
  int m0 = (wave >> 3) * 16;
  const _Float16* arow = A + (long)(m0 + l15) * DINNER;
  const _Float16* b0 = Bt + (long)(n0 + 0  + l15) * DINNER;
  const _Float16* b1 = Bt + (long)(n0 + 16 + l15) * DINNER;
  const _Float16* b2 = Bt + (long)(n0 + 32 + l15) * DINNER;
  const _Float16* b3 = Bt + (long)(n0 + 48 + l15) * DINNER;
  v8f c0 = {}, c1 = {}, c2 = {}, c3 = {};
  mm_strip<DINNER>(arow, b0, b1, b2, b3, lhi, c0, c1, c2, c3);
#pragma unroll
  for (int r = 0; r < 8; ++r) {
    long row = m0 + r + lhi * 8;
    long i0 = row * DMODEL + n0 + l15;
    preLN[i0 + 0 ] = c0[r] + bias[n0 + 0  + l15] + resid[i0 + 0 ];
    preLN[i0 + 16] = c1[r] + bias[n0 + 16 + l15] + resid[i0 + 16];
    preLN[i0 + 32] = c2[r] + bias[n0 + 32 + l15] + resid[i0 + 32];
    preLN[i0 + 48] = c3[r] + bias[n0 + 48 + l15] + resid[i0 + 48];
  }
}

// ---------------- launch ----------------
extern "C" void kernel_launch(void* const* d_in, const int* in_sizes, int n_in,
                              void* d_out, int out_size, void* d_ws, size_t ws_size,
                              hipStream_t stream) {
  const float* x      = (const float*)d_in[0];
  const float* w_qs   = (const float*)d_in[1];
  const float* w_ks   = (const float*)d_in[2];
  const float* w_vs   = (const float*)d_in[3];
  const float* proj_w = (const float*)d_in[4];
  const float* proj_b = (const float*)d_in[5];
  const float* ln1_g  = (const float*)d_in[6];
  const float* ln1_b  = (const float*)d_in[7];
  const float* ffn_w1 = (const float*)d_in[8];
  const float* ffn_b1 = (const float*)d_in[9];
  const float* ffn_w2 = (const float*)d_in[10];
  const float* ffn_b2 = (const float*)d_in[11];
  const float* ln2_g  = (const float*)d_in[12];
  const float* ln2_b  = (const float*)d_in[13];
  const int*   keep   = (const int*)d_in[14];

  float* enc_out = (float*)d_out;              // [B,S,D]
  float* attns   = (float*)d_out + ENC_ELEMS;  // [H*B,S,S]

  // --- workspace carve-out (256B aligned) ---
  char* base = (char*)d_ws;
  size_t off = 0;
  auto alloc = [&](size_t bytes) -> void* {
    void* p = base + off;
    off = (off + bytes + 255) & ~(size_t)255;
    return p;
  };
  int*       srcidx   = (int*)alloc((size_t)MROWS * 4);
  int*       nkeep    = (int*)alloc(BB * 4);
  float*     xc       = (float*)alloc((size_t)MROWS * DMODEL * 4);      // 32 MB
  _Float16*  xch      = (_Float16*)alloc((size_t)MROWS * DMODEL * 2);   // 16 MB
  _Float16*  wqT      = (_Float16*)alloc((size_t)DMODEL * DMODEL * 2);
  _Float16*  wkT      = (_Float16*)alloc((size_t)DMODEL * DMODEL * 2);
  _Float16*  wvT      = (_Float16*)alloc((size_t)DMODEL * DMODEL * 2);
  _Float16*  projT    = (_Float16*)alloc((size_t)DMODEL * DMODEL * 2);
  _Float16*  w1T      = (_Float16*)alloc((size_t)DMODEL * DINNER * 2);
  _Float16*  w2T      = (_Float16*)alloc((size_t)DMODEL * DINNER * 2);
  _Float16*  qh       = (_Float16*)alloc((size_t)MROWS * DMODEL * 2);   // 16 MB
  _Float16*  kh       = (_Float16*)alloc((size_t)MROWS * DMODEL * 2);   // 16 MB
  _Float16*  vT       = (_Float16*)alloc((size_t)MROWS * DMODEL * 2);   // 16 MB
  _Float16*  heads    = (_Float16*)alloc((size_t)MROWS * DMODEL * 2);   // 16 MB
  float*     preLN    = (float*)alloc((size_t)MROWS * DMODEL * 4);      // 32 MB (reused)
  float*     attnout  = (float*)alloc((size_t)MROWS * DMODEL * 4);      // 32 MB
  _Float16*  attnouth = (_Float16*)alloc((size_t)MROWS * DMODEL * 2);   // 16 MB
  _Float16*  hbuf     = (_Float16*)alloc((size_t)MROWS * DINNER * 2);   // 64 MB
  (void)ws_size; (void)in_sizes; (void)n_in; (void)out_size;

  // 1. pack weights (transpose + fp32->f16)
  k_pack_qkv<<<(DMODEL * DMODEL + 255) / 256, 256, 0, stream>>>(w_qs, wqT);
  k_pack_qkv<<<(DMODEL * DMODEL + 255) / 256, 256, 0, stream>>>(w_ks, wkT);
  k_pack_qkv<<<(DMODEL * DMODEL + 255) / 256, 256, 0, stream>>>(w_vs, wvT);
  k_pack_T<<<(DMODEL * DMODEL + 255) / 256, 256, 0, stream>>>(proj_w, projT, DMODEL, DMODEL);
  k_pack_T<<<(DMODEL * DINNER + 255) / 256, 256, 0, stream>>>(ffn_w1, w1T, DINNER, DMODEL);
  k_pack_T<<<(DMODEL * DINNER + 255) / 256, 256, 0, stream>>>(ffn_w2, w2T, DMODEL, DINNER);

  // 2. skipper: compact kept timesteps, zero-pad
  k_compact<<<BB, SS, 0, stream>>>(keep, srcidx, nkeep);
  k_gather<<<MROWS, 128, 0, stream>>>(x, srcidx, nkeep, xc, xch);

  // 3. QKV projections (8192 waves each: 1024 m-tiles x 8 strips)
  k_gemm_qk<<<1024, 256, 0, stream>>>(xch, wqT, qh);
  k_gemm_qk<<<1024, 256, 0, stream>>>(xch, wkT, kh);
  k_gemm_v <<<1024, 256, 0, stream>>>(xch, wvT, vT);

  // 4. scores -> d_out attns region (65536 waves)
  k_scores<<<8192, 256, 0, stream>>>(qh, kh, attns);

  // 5. masked softmax in place
  k_softmax<<<NHEAD * BB * SS, SS, 0, stream>>>(attns, nkeep);

  // 6. attn @ V -> heads (8192 waves)
  k_attnv<<<1024, 256, 0, stream>>>(attns, vT, heads);

  // 7. proj + bias + residual -> LN1 (fp32 + f16 shadow)
  k_proj<<<1024, 256, 0, stream>>>(heads, projT, proj_b, xc, preLN);
  k_layernorm<<<MROWS, DMODEL, 0, stream>>>(preLN, ln1_g, ln1_b, attnout, attnouth);

  // 8. FFN
  k_ffn1<<<4096, 256, 0, stream>>>(attnouth, w1T, ffn_b1, hbuf);
  k_ffn2<<<1024, 256, 0, stream>>>(hbuf, w2T, ffn_b2, attnout, preLN);
  k_layernorm<<<MROWS, DMODEL, 0, stream>>>(preLN, ln2_g, ln2_b, enc_out, (_Float16*)nullptr);
}
